// FlexibleJSSNet_16252156248139
// MI455X (gfx1250) — matrix operand order repaired and checked by reading, not compiled, not run on hardware
//
#include <hip/hip_runtime.h>

// ---------------- CDNA5 WMMA helpers (wave32, 16x16x32 f16 -> f32) ----------

typedef __attribute__((ext_vector_type(16))) _Float16 v16h;
typedef __attribute__((ext_vector_type(8)))  _Float16 v8h;
typedef __attribute__((ext_vector_type(8)))  float    v8f;

#define LDS_FENCE() asm volatile("s_wait_dscnt 0x0" ::: "memory")

__device__ __forceinline__ v8f wmma_f16(v16h a, v16h b, v8f c) {
  // (neg_a, A, neg_b, B, c_mod, C, reuse_a, reuse_b)
  return __builtin_amdgcn_wmma_f32_16x16x32_f16(false, a, false, b, (short)0, c,
                                                false, false);
}

// Load a 16x32 A-tile from a row-major f16 LDS buffer as two contiguous
// 8-half chunks per lane (-> ds_load_b128 x2).
// A layout: lanes 0-15 = rows M, K = {kb..kb+7, kb+16..kb+23}, kb = 8*(lane>>4).
__device__ __forceinline__ v16h load_a16(const _Float16* base, int lda, int k0,
                                         int lane) {
  int m  = lane & 15;
  int kb = (lane >> 4) << 3;
  const _Float16* p = base + m * lda + k0 + kb;
  v8h lo = *(const v8h*)(p);
  v8h hi = *(const v8h*)(p + 16);
  return __builtin_shufflevector(lo, hi, 0, 1, 2, 3, 4, 5, 6, 7,
                                 8, 9, 10, 11, 12, 13, 14, 15);
}

// Load a pre-packed 32x16 B fragment: 16 contiguous halfs per lane.
__device__ __forceinline__ v16h load_bp(const _Float16* pk, int frag, int lane) {
  return *(const v16h*)(pk + (frag * 32 + lane) * 16);
}

// Store a 16x16 f32 C tile into a row-major f16 LDS buffer, + bias, opt. relu.
// C/D layout: lane = N, VGPR i = row i (+8 for upper half-wave).
__device__ __forceinline__ void store_c16(_Float16* base, int ld, int n0, v8f c,
                                          const float* bias, bool relu, int lane) {
  int n  = (lane & 15) + n0;
  int mo = (lane >> 4) << 3;
  float bv = bias[n];
#pragma unroll
  for (int i = 0; i < 8; ++i) {
    float v = c[i] + bv;
    if (relu) v = fmaxf(v, 0.0f);
    base[(mo + i) * ld + n] = (_Float16)v;
  }
}

// ---------------- problem constants -----------------------------------------

#define B_GRAPHS 64
#define MPG      32
#define TPG      8192
#define NM       (B_GRAPHS * MPG)   // 2048
#define NT       (B_GRAPHS * TPG)   // 524288
#define NL       65536
#define DT       18
#define DM       8
#define H        64

// Packed-weight fragment offsets (in halfs). Each fragment = 32 lanes * 16 = 512.
#define PK_TW1   0          // 1 kstep  * 4 = 2048
#define PK_TW2   2048       // 2 ksteps * 4 = 4096
#define PK_MW1   6144       // 1 * 4
#define PK_MW2   8192       // 2 * 4
#define PK_AW1   12288      // 4 * 4
#define PK_AW2   20480      // 2 * 4
#define PK_LGW1  24576      // 6 * 4
#define PK_LGW2  36864      // 2 * 4
#define PK_OW1   40960      // 2 * 4
#define PK_TOTAL 45056

// ---------------- kernels ----------------------------------------------------

__global__ __launch_bounds__(256) void zero_kernel(float* p, int n) {
  int i = blockIdx.x * blockDim.x + threadIdx.x;
  if (i < n) p[i] = 0.0f;
}

// Pack a f32 weight matrix W[K][64] into f16 B fragments (zero-padded K<kmax).
// grid = nks*4 fragments, 512 threads = one element each.
__global__ __launch_bounds__(512) void pack_w(const float* __restrict__ W,
                                              _Float16* __restrict__ dst, int kmax) {
  int frag = blockIdx.x;            // ks*4 + nt
  int e    = threadIdx.x;           // lane*16 + i
  int lane = e >> 4, i = e & 15;
  int ks = frag >> 2, nt = frag & 3;
  int n = (lane & 15) + nt * 16;
  int k = ks * 32 + ((lane >> 4) << 3) + (i & 7) + ((i >> 3) << 4);
  dst[frag * 512 + e] = (k < kmax) ? (_Float16)W[k * H + n] : (_Float16)0.0f;
}

// Machines: LN(8) -> 8->64 relu -> 64->64. Writes mh16[NM][64] (f16) + graph sums.
// One wave per 16-machine tile, 8 waves/block, 128 tiles.
__global__ __launch_bounds__(256) void machine_kernel(
    const float* __restrict__ xm, const float* __restrict__ ln_g,
    const float* __restrict__ ln_b,
    const _Float16* __restrict__ pk,
    const float* __restrict__ b1, const float* __restrict__ b2,
    _Float16* __restrict__ mh16, float* __restrict__ mh_sum) {
  __shared__ float    s_raw[8][16 * DM];
  __shared__ _Float16 s_xn[8][16 * 32];
  __shared__ _Float16 s_h1[8][16 * 64];
  int wave = threadIdx.x >> 5;
  int lane = threadIdx.x & 31;
  int tile = blockIdx.x * 8 + wave;
  float*    raw = s_raw[wave];
  _Float16* xn  = s_xn[wave];
  _Float16* h1  = s_h1[wave];

  // coalesced stage of the 16x8 input tile
  for (int idx = lane; idx < 16 * DM; idx += 32) raw[idx] = xm[tile * 16 * DM + idx];
  LDS_FENCE();

  if (lane < 16) {
    float v[DM], mu = 0.f;
#pragma unroll
    for (int j = 0; j < DM; ++j) { v[j] = raw[lane * DM + j]; mu += v[j]; }
    mu *= (1.0f / DM);
    float var = 0.f;
#pragma unroll
    for (int j = 0; j < DM; ++j) { float d = v[j] - mu; var += d * d; }
    float rs = rsqrtf(var * (1.0f / DM) + 1e-5f);
#pragma unroll
    for (int j = 0; j < DM; ++j)
      xn[lane * 32 + j] = (_Float16)((v[j] - mu) * rs * ln_g[j] + ln_b[j]);
#pragma unroll
    for (int j = DM; j < 32; ++j) xn[lane * 32 + j] = (_Float16)0.0f;
  }
  LDS_FENCE();

  v16h a = load_a16(xn, 32, 0, lane);
#pragma unroll
  for (int nt = 0; nt < 4; ++nt) {
    v8f c = {};
    c = wmma_f16(a, load_bp(pk + PK_MW1, nt, lane), c);
    store_c16(h1, H, nt * 16, c, b1, true, lane);
  }
  LDS_FENCE();

  v16h a0 = load_a16(h1, H, 0, lane);
  v16h a1 = load_a16(h1, H, 32, lane);
  int graph = tile >> 1;
#pragma unroll
  for (int nt = 0; nt < 4; ++nt) {
    v8f c = {};
    c = wmma_f16(a0, load_bp(pk + PK_MW2, 0 * 4 + nt, lane), c);
    c = wmma_f16(a1, load_bp(pk + PK_MW2, 1 * 4 + nt, lane), c);
    int n  = (lane & 15) + nt * 16;
    int mo = (lane >> 4) << 3;
    float bv = b2[n];
    float acc = 0.f;
#pragma unroll
    for (int i = 0; i < 8; ++i) {
      float v = c[i] + bv;
      mh16[(tile * 16 + mo + i) * H + n] = (_Float16)v;
      acc += v;
    }
    atomicAdd(&mh_sum[graph * H + n], acc);
  }
}

// Tasks: LN(18) -> 18->64 relu -> 64->64; accumulate per-graph column sums only.
// 32768 tiles, 8 waves/block; all 8 tiles of a block share one graph, so sums
// are block-reduced in LDS before one global atomic per column.
__global__ __launch_bounds__(256) void task_kernel(
    const float* __restrict__ xt, const float* __restrict__ ln_g,
    const float* __restrict__ ln_b,
    const _Float16* __restrict__ pk,
    const float* __restrict__ b1, const float* __restrict__ b2,
    float* __restrict__ th_sum) {
  __shared__ float    s_raw[8][16 * DT];
  __shared__ _Float16 s_xn[8][16 * 32];
  __shared__ _Float16 s_h1[8][16 * 64];
  __shared__ float    red[H];
  int wave = threadIdx.x >> 5;
  int lane = threadIdx.x & 31;
  int tile = blockIdx.x * 8 + wave;
  float*    raw = s_raw[wave];
  _Float16* xn  = s_xn[wave];
  _Float16* h1  = s_h1[wave];

  if (threadIdx.x < H) red[threadIdx.x] = 0.f;
  __syncthreads();

  // coalesced stage of the 16x18 input tile (the dominant HBM stream)
  for (int idx = lane; idx < 16 * DT; idx += 32) raw[idx] = xt[tile * 16 * DT + idx];
  LDS_FENCE();

  if (lane < 16) {
    float v[DT], mu = 0.f;
#pragma unroll
    for (int j = 0; j < DT; ++j) { v[j] = raw[lane * DT + j]; mu += v[j]; }
    mu *= (1.0f / DT);
    float var = 0.f;
#pragma unroll
    for (int j = 0; j < DT; ++j) { float d = v[j] - mu; var += d * d; }
    float rs = rsqrtf(var * (1.0f / DT) + 1e-5f);
#pragma unroll
    for (int j = 0; j < DT; ++j)
      xn[lane * 32 + j] = (_Float16)((v[j] - mu) * rs * ln_g[j] + ln_b[j]);
#pragma unroll
    for (int j = DT; j < 32; ++j) xn[lane * 32 + j] = (_Float16)0.0f;
  }
  LDS_FENCE();

  v16h a = load_a16(xn, 32, 0, lane);
#pragma unroll
  for (int nt = 0; nt < 4; ++nt) {
    v8f c = {};
    c = wmma_f16(a, load_bp(pk + PK_TW1, nt, lane), c);
    store_c16(h1, H, nt * 16, c, b1, true, lane);
  }
  LDS_FENCE();

  v16h a0 = load_a16(h1, H, 0, lane);
  v16h a1 = load_a16(h1, H, 32, lane);
#pragma unroll
  for (int nt = 0; nt < 4; ++nt) {
    v8f c = {};
    c = wmma_f16(a0, load_bp(pk + PK_TW2, 0 * 4 + nt, lane), c);
    c = wmma_f16(a1, load_bp(pk + PK_TW2, 1 * 4 + nt, lane), c);
    int n  = (lane & 15) + nt * 16;
    float bv = b2[n];
    float acc = 0.f;
#pragma unroll
    for (int i = 0; i < 8; ++i) acc += c[i] + bv;
    atomicAdd(&red[n], acc);          // LDS ds_add_f32
  }
  __syncthreads();
  if (threadIdx.x < H) {
    int graph = (blockIdx.x * 8) >> 9; // 64 blocks per graph
    atomicAdd(&th_sum[graph * H + threadIdx.x], red[threadIdx.x]);
  }
}

// Graph aggregation: [mean_th | mean_mh] (64x128) -> 128->64 relu -> 64->64.
// 4 single-wave blocks of 16 graphs each; counts are static (8192, 32).
__global__ __launch_bounds__(32) void aggr_kernel(
    const float* __restrict__ th_sum, const float* __restrict__ mh_sum,
    const _Float16* __restrict__ pk,
    const float* __restrict__ b1, const float* __restrict__ b2,
    _Float16* __restrict__ aggr16) {
  __shared__ _Float16 cat[16 * 128];
  __shared__ _Float16 h1[16 * 64];
  int lane = threadIdx.x;
  int g0 = blockIdx.x * 16;

  for (int idx = lane; idx < 16 * 128; idx += 32) {
    int m = idx >> 7, j = idx & 127;
    float v = (j < H) ? th_sum[(g0 + m) * H + j] * (1.0f / TPG)
                      : mh_sum[(g0 + m) * H + (j - H)] * (1.0f / MPG);
    cat[idx] = (_Float16)v;
  }
  LDS_FENCE();

#pragma unroll
  for (int nt = 0; nt < 4; ++nt) {
    v8f c = {};
#pragma unroll
    for (int ks = 0; ks < 4; ++ks)
      c = wmma_f16(load_a16(cat, 128, ks * 32, lane),
                   load_bp(pk + PK_AW1, ks * 4 + nt, lane), c);
    store_c16(h1, H, nt * 16, c, b1, true, lane);
  }
  LDS_FENCE();

  v16h a0 = load_a16(h1, H, 0, lane);
  v16h a1 = load_a16(h1, H, 32, lane);
#pragma unroll
  for (int nt = 0; nt < 4; ++nt) {
    v8f c = {};
    c = wmma_f16(a0, load_bp(pk + PK_AW2, 0 * 4 + nt, lane), c);
    c = wmma_f16(a1, load_bp(pk + PK_AW2, 1 * 4 + nt, lane), c);
    int n  = (lane & 15) + nt * 16;
    int mo = (lane >> 4) << 3;
    float bv = b2[n];
#pragma unroll
    for (int i = 0; i < 8; ++i)
      aggr16[(g0 + mo + i) * H + n] = (_Float16)(c[i] + bv);
  }
}

// Label head: for the 65536 labeled tasks only — recompute th (cheap), gather
// aggr16[graph] and mh16[machine], run 192->64->64 then 64->64->1.
// One wave per 16-label tile, 4 waves/block, 4096 tiles.
__global__ __launch_bounds__(128) void label_kernel(
    const float* __restrict__ xt, const int* __restrict__ labels,
    const float* __restrict__ t_ln_g, const float* __restrict__ t_ln_b,
    const _Float16* __restrict__ pk,
    const float* __restrict__ t_b1, const float* __restrict__ t_b2,
    const _Float16* __restrict__ mh16, const _Float16* __restrict__ aggr16,
    const float* __restrict__ lg_b1, const float* __restrict__ lg_b2,
    const float* __restrict__ o_b1,
    const float* __restrict__ o_w2, const float* __restrict__ o_b2,
    float* __restrict__ out) {
  __shared__ _Float16 s_cat[4][16 * 192];
  __shared__ _Float16 s_h[4][16 * 64];
  __shared__ int      s_g[4][16];
  __shared__ int      s_m[4][16];
  int wave = threadIdx.x >> 5;
  int lane = threadIdx.x & 31;
  int tile = blockIdx.x * 4 + wave;
  _Float16* cat = s_cat[wave];
  _Float16* h   = s_h[wave];
  int* sg = s_g[wave];
  int* sm = s_m[wave];

  if (lane < 16) {
    int t = labels[tile * 16 + lane];
    float v[DT], mu = 0.f;
#pragma unroll
    for (int j = 0; j < DT; ++j) { v[j] = xt[t * DT + j]; mu += v[j]; }
    mu *= (1.0f / DT);
    float var = 0.f;
#pragma unroll
    for (int j = 0; j < DT; ++j) { float d = v[j] - mu; var += d * d; }
    float rs = rsqrtf(var * (1.0f / DT) + 1e-5f);
#pragma unroll
    for (int j = 0; j < DT; ++j)
      cat[lane * 192 + j] = (_Float16)((v[j] - mu) * rs * t_ln_g[j] + t_ln_b[j]);
#pragma unroll
    for (int j = DT; j < 32; ++j) cat[lane * 192 + j] = (_Float16)0.0f;

    int graph  = t >> 13;                 // t / 8192
    int assign = (int)v[0];               // raw feature 0, -1 = unassigned
    sg[lane] = graph;
    sm[lane] = (assign < 0) ? -1 : (graph * MPG + assign);
  }
  LDS_FENCE();

  // cooperative dword gather of aggr/mh rows into cat cols 64..191
  unsigned int* catu = (unsigned int*)cat;
#pragma unroll 4
  for (int r = 0; r < 16; ++r) {
    int g = sg[r], m = sm[r];
    catu[((r * 192 + 64) >> 1) + lane] =
        ((const unsigned int*)(aggr16 + g * H))[lane];
    unsigned int mv = 0u;
    if (m >= 0) mv = ((const unsigned int*)(mh16 + m * H))[lane];
    catu[((r * 192 + 128) >> 1) + lane] = mv;
  }
  LDS_FENCE();

  // th layer1 (K=18 pre-padded in weights, zeros in cat cols 18..31) -> h
  v16h a = load_a16(cat, 192, 0, lane);
#pragma unroll
  for (int nt = 0; nt < 4; ++nt) {
    v8f c = {};
    c = wmma_f16(a, load_bp(pk + PK_TW1, nt, lane), c);
    store_c16(h, H, nt * 16, c, t_b1, true, lane);
  }
  LDS_FENCE();

  // th layer2 -> cat cols 0..63
  v16h a0 = load_a16(h, H, 0, lane);
  v16h a1 = load_a16(h, H, 32, lane);
#pragma unroll
  for (int nt = 0; nt < 4; ++nt) {
    v8f c = {};
    c = wmma_f16(a0, load_bp(pk + PK_TW2, 0 * 4 + nt, lane), c);
    c = wmma_f16(a1, load_bp(pk + PK_TW2, 1 * 4 + nt, lane), c);
    store_c16(cat, 192, nt * 16, c, t_b2, false, lane);
  }
  LDS_FENCE();

  // lg layer1: K=192 (6 k-steps) -> h (relu)
#pragma unroll
  for (int nt = 0; nt < 4; ++nt) {
    v8f c = {};
#pragma unroll
    for (int ks = 0; ks < 6; ++ks)
      c = wmma_f16(load_a16(cat, 192, ks * 32, lane),
                   load_bp(pk + PK_LGW1, ks * 4 + nt, lane), c);
    store_c16(h, H, nt * 16, c, lg_b1, true, lane);
  }
  LDS_FENCE();

  // lg layer2 -> hid (reuse cat storage, ld=64), no relu
  a0 = load_a16(h, H, 0, lane);
  a1 = load_a16(h, H, 32, lane);
  _Float16* hid = cat;
#pragma unroll
  for (int nt = 0; nt < 4; ++nt) {
    v8f c = {};
    c = wmma_f16(a0, load_bp(pk + PK_LGW2, 0 * 4 + nt, lane), c);
    c = wmma_f16(a1, load_bp(pk + PK_LGW2, 1 * 4 + nt, lane), c);
    store_c16(hid, H, nt * 16, c, lg_b2, false, lane);
  }
  LDS_FENCE();

  // o layer1 -> h (relu)
  a0 = load_a16(hid, H, 0, lane);
  a1 = load_a16(hid, H, 32, lane);
#pragma unroll
  for (int nt = 0; nt < 4; ++nt) {
    v8f c = {};
    c = wmma_f16(a0, load_bp(pk + PK_OW1, 0 * 4 + nt, lane), c);
    c = wmma_f16(a1, load_bp(pk + PK_OW1, 1 * 4 + nt, lane), c);
    store_c16(h, H, nt * 16, c, o_b1, true, lane);
  }
  LDS_FENCE();

  // final 64->1 dot per row
  if (lane < 16) {
    float s = o_b2[0];
#pragma unroll
    for (int k = 0; k < H; ++k) s += (float)h[lane * H + k] * o_w2[k];
    out[tile * 16 + lane] = s;
  }
}

// ---------------- host launch -----------------------------------------------

extern "C" void kernel_launch(void* const* d_in, const int* in_sizes, int n_in,
                              void* d_out, int out_size, void* d_ws, size_t ws_size,
                              hipStream_t stream) {
  const float* x_tasks    = (const float*)d_in[0];
  const float* x_machines = (const float*)d_in[1];
  // d_in[2], d_in[3]: batch indices — statically known (i//8192, i//32), unused.
  const int*   labels     = (const int*)d_in[4];
  const float* t_ln_g = (const float*)d_in[5];
  const float* t_ln_b = (const float*)d_in[6];
  const float* m_ln_g = (const float*)d_in[7];
  const float* m_ln_b = (const float*)d_in[8];
  const float* t_w1   = (const float*)d_in[9];
  const float* t_b1   = (const float*)d_in[10];
  const float* t_w2   = (const float*)d_in[11];
  const float* t_b2   = (const float*)d_in[12];
  const float* m_w1   = (const float*)d_in[13];
  const float* m_b1   = (const float*)d_in[14];
  const float* m_w2   = (const float*)d_in[15];
  const float* m_b2   = (const float*)d_in[16];
  const float* a_w1   = (const float*)d_in[17];
  const float* a_b1   = (const float*)d_in[18];
  const float* a_w2   = (const float*)d_in[19];
  const float* a_b2   = (const float*)d_in[20];
  const float* lg_w1  = (const float*)d_in[21];
  const float* lg_b1  = (const float*)d_in[22];
  const float* lg_w2  = (const float*)d_in[23];
  const float* lg_b2  = (const float*)d_in[24];
  const float* o_w1   = (const float*)d_in[25];
  const float* o_b1   = (const float*)d_in[26];
  const float* o_w2   = (const float*)d_in[27];
  const float* o_b2   = (const float*)d_in[28];

  // workspace layout
  float*    th_sum = (float*)d_ws;                    // 4096 f32
  float*    mh_sum = th_sum + B_GRAPHS * H;           // 4096 f32
  _Float16* aggr16 = (_Float16*)(mh_sum + B_GRAPHS * H);  // 4096 f16
  _Float16* mh16   = aggr16 + B_GRAPHS * H;           // 131072 f16
  _Float16* pk     = mh16 + NM * H;                   // 45056 f16 packed weights

  // zero the accumulators
  zero_kernel<<<(2 * B_GRAPHS * H + 255) / 256, 256, 0, stream>>>(
      th_sum, 2 * B_GRAPHS * H);

  // pack all 64-column weight matrices into f16 B fragments (zero K-padding)
  pack_w<<<1 * 4, 512, 0, stream>>>(t_w1,  pk + PK_TW1,  DT);
  pack_w<<<2 * 4, 512, 0, stream>>>(t_w2,  pk + PK_TW2,  H);
  pack_w<<<1 * 4, 512, 0, stream>>>(m_w1,  pk + PK_MW1,  DM);
  pack_w<<<2 * 4, 512, 0, stream>>>(m_w2,  pk + PK_MW2,  H);
  pack_w<<<4 * 4, 512, 0, stream>>>(a_w1,  pk + PK_AW1,  128);
  pack_w<<<2 * 4, 512, 0, stream>>>(a_w2,  pk + PK_AW2,  H);
  pack_w<<<6 * 4, 512, 0, stream>>>(lg_w1, pk + PK_LGW1, 192);
  pack_w<<<2 * 4, 512, 0, stream>>>(lg_w2, pk + PK_LGW2, H);
  pack_w<<<2 * 4, 512, 0, stream>>>(o_w1,  pk + PK_OW1,  H);

  machine_kernel<<<NM / 16 / 8, 256, 0, stream>>>(
      x_machines, m_ln_g, m_ln_b, pk, m_b1, m_b2, mh16, mh_sum);
  task_kernel<<<NT / 16 / 8, 256, 0, stream>>>(
      x_tasks, t_ln_g, t_ln_b, pk, t_b1, t_b2, th_sum);
  aggr_kernel<<<B_GRAPHS / 16, 32, 0, stream>>>(
      th_sum, mh_sum, pk, a_b1, a_b2, aggr16);
  label_kernel<<<NL / 16 / 4, 128, 0, stream>>>(
      x_tasks, labels, t_ln_g, t_ln_b, pk, t_b1, t_b2, mh16, aggr16,
      lg_b1, lg_b2, o_b1, o_w2, o_b2, (float*)d_out);
}
